// ImplicitEGNN_44796508897961
// MI455X (gfx1250) — matrix operand order repaired and checked by reading, not compiled, not run on hardware
//
#include <hip/hip_runtime.h>
#include <hip/hip_bf16.h>

// ---------------------------------------------------------------------------
// Problem constants (fixed by the reference)
// ---------------------------------------------------------------------------
#define NN 10000      // nodes  (multiple of 16)
#define EE 160000     // edges  (multiple of 32)
#define NITER 8

typedef __bf16 bf16;
typedef __attribute__((ext_vector_type(16))) __bf16 v16bf;
typedef __attribute__((ext_vector_type(8)))  float  v8f;

// ---------------------------------------------------------------------------
// Helpers
// ---------------------------------------------------------------------------
__device__ __forceinline__ float gelu_tanh(float x) {
    const float k0 = 0.7978845608028654f;   // sqrt(2/pi)
    const float k1 = 0.044715f;
    float t = tanhf(k0 * (x + k1 * x * x * x));
    return 0.5f * x * (1.0f + t);
}

__device__ __forceinline__ v8f wmma_bf16(v16bf a, v16bf b, v8f c) {
    // D = A(16x32 bf16) x B(32x16 bf16) + C(16x16 f32)
    return __builtin_amdgcn_wmma_f32_16x16x32_bf16(
        false, a, false, b, (short)0, c, false, false);
}

// A-fragment: 16x32 bf16 tile from row-major LDS buffer, row stride ldm elems.
// Lane l holds row M=l&15; lanes 0-15 carry K 0..15,
// lanes 16-31 carry K 16..31; consecutive elements = consecutive K.
__device__ __forceinline__ v16bf load_a_frag(const bf16* smem, int ldm, int k0) {
    int lane = threadIdx.x & 31;
    int m    = lane & 15;
    int kh   = (lane >> 4) << 4;          // 0 or 16
    const bf16* p = smem + m * ldm + k0;
    v16bf r;
#pragma unroll
    for (int i = 0; i < 8; ++i) r[i]     = p[kh + i];        // K = kh..kh+7
#pragma unroll
    for (int i = 0; i < 8; ++i) r[8 + i] = p[8 + kh + i];    // K = kh+8..kh+15
    return r;
}

// B-fragment: 32x16 tile of W^T stored [Ncols][K] row-major (ldk elems/row).
// Lane l holds column N=l&15; lanes 0-15 -> K 0..15, lanes 16-31 -> K 16..31.
// 16 contiguous bf16 per lane = 32B aligned loads.
__device__ __forceinline__ v16bf load_b_frag(const bf16* Bt, int ldk, int n0, int k0) {
    int lane = threadIdx.x & 31;
    int n    = lane & 15;
    int kh   = (lane >> 4) << 4;          // 0 or 16
    const bf16* p = Bt + (size_t)(n0 + n) * ldk + k0 + kh;
    v16bf r;
#pragma unroll
    for (int i = 0; i < 16; ++i) r[i] = p[i];
    return r;
}

// ---------------------------------------------------------------------------
// One-time prep kernels
// ---------------------------------------------------------------------------
__global__ void k_zero(float* __restrict__ p, int n) {
    int i = blockIdx.x * blockDim.x + threadIdx.x;
    if (i < n) p[i] = 0.0f;
}

// src[R][C] f32 -> dst[C][R] bf16 (transpose + convert)
__global__ void k_transpose_bf16(const float* __restrict__ src, bf16* __restrict__ dst,
                                 int R, int C) {
    int i = blockIdx.x * blockDim.x + threadIdx.x;
    if (i >= R * C) return;
    int r = i / C, c = i % C;
    dst[(size_t)c * R + r] = (bf16)src[(size_t)r * C + c];
}

// Build transposed [512][128] bf16 of [we1_a | we1_b] from we1 [273][256]
__global__ void k_build_we1ab_t(const float* __restrict__ we1, bf16* __restrict__ bt) {
    int i = blockIdx.x * blockDim.x + threadIdx.x;
    if (i >= 512 * 128) return;
    int n = i >> 7, k = i & 127;
    float v = (n < 256) ? we1[(size_t)k * 256 + n]
                        : we1[(size_t)(k + 128) * 256 + (n - 256)];
    bt[i] = (bf16)v;
}

// Wcd [18][256] f32 = rows [be1 ; we1 row 256 (d2) ; we1 rows 257..272 (ea)]
__global__ void k_build_wcd(const float* __restrict__ we1, const float* __restrict__ be1,
                            float* __restrict__ wcd) {
    int i = blockIdx.x * blockDim.x + threadIdx.x;
    if (i >= 18 * 256) return;
    int r = i >> 8, c = i & 255;
    wcd[i] = (r == 0) ? be1[c] : we1[(size_t)(255 + r) * 256 + c];
}

// ---------------------------------------------------------------------------
// Row GEMM, K=128, compile-time NCOLS: out = A_f32[n][128] @ Bt^T + bias.
// 16 rows/block. B fragments batched (consumed in reverse so all stay live ->
// one clause of loads, one wait, 4 back-to-back WMMAs).
// ---------------------------------------------------------------------------
#define LDA 136   // 17*8 -> 16B-aligned chunks, conflict-friendly
template<int NCOLS>
__global__ __launch_bounds__(128)
void k_gemm_n16k128(const float* __restrict__ A, const bf16* __restrict__ Bt,
                    const float* __restrict__ bias, float* __restrict__ out) {
    __shared__ __align__(16) bf16 sA[16 * LDA];
    int row0 = blockIdx.x << 4;
    int tid  = threadIdx.x;                      // 128 threads, 4 waves
    for (int i = tid; i < 16 * 128; i += 128) {
        int r = i >> 7, c = i & 127;
        sA[r * LDA + c] = (bf16)A[(size_t)(row0 + r) * 128 + c];
    }
    __syncthreads();
    int wave = tid >> 5, lane = tid & 31;
    v16bf af[4];
#pragma unroll
    for (int k8 = 0; k8 < 4; ++k8) af[k8] = load_a_frag(sA, LDA, k8 << 5);
#pragma unroll
    for (int t = wave; t < (NCOLS >> 4); t += 4) {
        v16bf bf[4];
#pragma unroll
        for (int k8 = 0; k8 < 4; ++k8) bf[k8] = load_b_frag(Bt, 128, t << 4, k8 << 5);
        v8f acc = {};
#pragma unroll
        for (int k8 = 3; k8 >= 0; --k8) acc = wmma_bf16(af[k8], bf[k8], acc);
        int n  = (t << 4) + (lane & 15);
        int mb = (lane >> 4) << 3;
        float bv = bias[n];
        float* op = out + (size_t)(row0 + mb) * NCOLS + n;
#pragma unroll
        for (int r = 0; r < 8; ++r) op[r * NCOLS] = acc[r] + bv;
    }
}

// ---------------------------------------------------------------------------
// Fused edge kernel: 32 edges/block, 256 threads (8 waves).
//  h = Zw[src][:256] + Zw[dst][256:] + [1,d2,ea]@Wcd ; H = gelu(h) (bf16, LDS)
//  m = gelu(H @ we2 + be2) ; agg[dst] += m  (f32 atomics, L2-resident)
// Each wave owns one column tile (nt=wave): B fragments loaded ONCE (before
// the VALU pre-activation work, to hide latency), reused for both row tiles.
// ---------------------------------------------------------------------------
#define HLD 264   // 33*8 elems -> 528B rows, 16B aligned, spreads banks
__global__ __launch_bounds__(256)
void k_edge(const float* __restrict__ Zw,      // [N][512]
            const float* __restrict__ pos,     // [N][3]
            const int*   __restrict__ srcI,    // [E]
            const int*   __restrict__ dstI,    // [E]
            const float* __restrict__ eattr,   // [E][16]
            const float* __restrict__ Wcd,     // [18][256]
            const bf16*  __restrict__ we2t,    // [128][256] (we2^T)
            const float* __restrict__ be2,     // [128]
            float* __restrict__ agg) {         // [N][128] (atomic accumulate)
    __shared__ __align__(16) float sW[18 * 256];
    __shared__ __align__(16) bf16  sH[32 * HLD];
    __shared__ int   sSrc[32], sDst[32];
    __shared__ float sU[32 * 18];
    int e0  = blockIdx.x << 5;
    int tid = threadIdx.x;
    int wave = tid >> 5, lane = tid & 31;

    for (int i = tid; i < 18 * 256; i += 256) sW[i] = Wcd[i];
    if (tid < 32) {
        int e = e0 + tid;
        int s = srcI[e], d = dstI[e];
        sSrc[tid] = s; sDst[tid] = d;
        float dx = pos[s * 3 + 0] - pos[d * 3 + 0];
        float dy = pos[s * 3 + 1] - pos[d * 3 + 1];
        float dz = pos[s * 3 + 2] - pos[d * 3 + 2];
        sU[tid * 18 + 0] = 1.0f;                         // bias lane
        sU[tid * 18 + 1] = dx * dx + dy * dy + dz * dz;  // d2
    }
    for (int i = tid; i < 32 * 16; i += 256) {
        int le = i >> 4, c = i & 15;
        sU[le * 18 + 2 + c] = eattr[(size_t)(e0 + le) * 16 + c];
    }

    // Issue this wave's 8 B fragments now; loads complete under the VALU work.
    v16bf bf[8];
#pragma unroll
    for (int k8 = 0; k8 < 8; ++k8) bf[k8] = load_b_frag(we2t, 256, wave << 4, k8 << 5);

    __syncthreads();

    // Pre-activation + gelu: 8 threads per edge, 32 cols each (coalesced L2 gathers)
    {
        int le = tid >> 3;
        int c0 = (tid & 7) << 5;
        int s = sSrc[le], d = sDst[le];
        const float* za = Zw + (size_t)s * 512;          // z[src] @ we1_a
        const float* zb = Zw + (size_t)d * 512 + 256;    // z[dst] @ we1_b
        float u[18];
#pragma unroll
        for (int i = 0; i < 18; ++i) u[i] = sU[le * 18 + i];
#pragma unroll
        for (int c = 0; c < 32; ++c) {
            int j = c0 + c;
            float acc = za[j] + zb[j];
#pragma unroll
            for (int i = 0; i < 18; ++i) acc = fmaf(u[i], sW[i * 256 + j], acc);
            sH[le * HLD + j] = (bf16)gelu_tanh(acc);
        }
    }
    __syncthreads();

    // m[32][128] = gelu(H @ we2 + be2); B reused across both 16-row tiles.
    v8f acc0 = {}, acc1 = {};
#pragma unroll
    for (int k8 = 0; k8 < 8; ++k8) {
        v16bf a0 = load_a_frag(sH,             HLD, k8 << 5);
        v16bf a1 = load_a_frag(sH + 16 * HLD,  HLD, k8 << 5);
        acc0 = wmma_bf16(a0, bf[k8], acc0);
        acc1 = wmma_bf16(a1, bf[k8], acc1);
    }
    int n  = (wave << 4) + (lane & 15);
    int mb = (lane >> 4) << 3;
    float bv = be2[n];
#pragma unroll
    for (int r = 0; r < 8; ++r) {
        int le = mb + r;
        atomicAdd(&agg[(size_t)sDst[le] * 128 + n], gelu_tanh(acc0[r] + bv));
    }
#pragma unroll
    for (int r = 0; r < 8; ++r) {
        int le = 16 + mb + r;
        atomicAdd(&agg[(size_t)sDst[le] * 128 + n], gelu_tanh(acc1[r] + bv));
    }
}

// ---------------------------------------------------------------------------
// Fused node kernel: 16 nodes/block, 128 threads (4 waves).
//  n_in = [z | agg] ; h = gelu(n_in@wn1+bn1) ; out = h@wn2+bn2
//  zout = gelu(out + resid)
// A fragments loaded once per GEMM, reused across all column tiles; B batched
// and consumed in reverse (clause -> single wait -> 8 WMMAs).
// ---------------------------------------------------------------------------
#define NLD 264
__global__ __launch_bounds__(128)
void k_node(const float* __restrict__ zin,   // [N][128]
            const float* __restrict__ agg,   // [N][128]
            const bf16*  __restrict__ wn1t,  // [256][256]
            const float* __restrict__ bn1,   // [256]
            const bf16*  __restrict__ wn2t,  // [128][256]
            const float* __restrict__ bn2,   // [128]
            const float* __restrict__ resid, // [N][128] (xin or z0)
            float* __restrict__ zout) {      // [N][128]
    __shared__ __align__(16) bf16 sA[16 * NLD];
    __shared__ __align__(16) bf16 sHn[16 * NLD];
    int row0 = blockIdx.x << 4;
    int tid  = threadIdx.x;
    for (int i = tid; i < 16 * 256; i += 128) {
        int r = i >> 8, c = i & 255;
        float v = (c < 128) ? zin[(size_t)(row0 + r) * 128 + c]
                            : agg[(size_t)(row0 + r) * 128 + (c - 128)];
        sA[r * NLD + c] = (bf16)v;
    }
    __syncthreads();
    int wave = tid >> 5, lane = tid & 31;
    int nl = lane & 15, mb = (lane >> 4) << 3;

    {   // hidden = gelu(n_in @ wn1 + bn1): 16 column tiles, 4 per wave
        v16bf af[8];
#pragma unroll
        for (int k8 = 0; k8 < 8; ++k8) af[k8] = load_a_frag(sA, NLD, k8 << 5);
#pragma unroll
        for (int t = 0; t < 4; ++t) {
            int nt = wave + (t << 2);
            v16bf bf[8];
#pragma unroll
            for (int k8 = 0; k8 < 8; ++k8)
                bf[k8] = load_b_frag(wn1t, 256, nt << 4, k8 << 5);
            v8f acc = {};
#pragma unroll
            for (int k8 = 7; k8 >= 0; --k8) acc = wmma_bf16(af[k8], bf[k8], acc);
            int n = (nt << 4) + nl;
            float bv = bn1[n];
#pragma unroll
            for (int r = 0; r < 8; ++r)
                sHn[(mb + r) * NLD + n] = (bf16)gelu_tanh(acc[r] + bv);
        }
    }
    __syncthreads();
    {   // out = hidden @ wn2 + bn2 ; zout = gelu(out + resid): 8 tiles, 2/wave
        v16bf af[8];
#pragma unroll
        for (int k8 = 0; k8 < 8; ++k8) af[k8] = load_a_frag(sHn, NLD, k8 << 5);
#pragma unroll
        for (int t = 0; t < 2; ++t) {
            int nt = wave + (t << 2);
            v16bf bf[8];
#pragma unroll
            for (int k8 = 0; k8 < 8; ++k8)
                bf[k8] = load_b_frag(wn2t, 256, nt << 4, k8 << 5);
            v8f acc = {};
#pragma unroll
            for (int k8 = 7; k8 >= 0; --k8) acc = wmma_bf16(af[k8], bf[k8], acc);
            int n = (nt << 4) + nl;
            float bv = bn2[n];
            const float* rp = resid + (size_t)(row0 + mb) * 128 + n;
            float*       zp = zout  + (size_t)(row0 + mb) * 128 + n;
#pragma unroll
            for (int r = 0; r < 8; ++r)
                zp[r * 128] = gelu_tanh(acc[r] + bv + rp[r * 128]);
        }
    }
}

// ---------------------------------------------------------------------------
// Host orchestration
// ---------------------------------------------------------------------------
extern "C" void kernel_launch(void* const* d_in, const int* in_sizes, int n_in,
                              void* d_out, int out_size, void* d_ws, size_t ws_size,
                              hipStream_t stream) {
    const float* x     = (const float*)d_in[0];
    const float* pos   = (const float*)d_in[1];
    const int*   eidx  = (const int*)d_in[2];
    const float* eattr = (const float*)d_in[3];
    const float* inj_w = (const float*)d_in[4];
    const float* inj_b = (const float*)d_in[5];
    const float* W[2][8];   // per layer: we1 be1 we2 be2 wn1 bn1 wn2 bn2
    for (int l = 0; l < 2; ++l)
        for (int j = 0; j < 8; ++j)
            W[l][j] = (const float*)d_in[6 + l * 8 + j];
    const int* srcI = eidx;
    const int* dstI = eidx + EE;

    // Workspace carve (256B aligned)
    char* base = (char*)d_ws;
    size_t off = 0;
    auto carve = [&](size_t bytes) {
        void* p = base + off;
        off += (bytes + 255) & ~(size_t)255;
        return p;
    };
    float* zZ    = (float*)carve((size_t)NN * 128 * 4);  // z0 / z_next
    float* z2    = (float*)carve((size_t)NN * 128 * 4);  // mid-iteration state
    float* xin   = (float*)carve((size_t)NN * 128 * 4);
    float* agg   = (float*)carve((size_t)NN * 128 * 4);
    float* zw    = (float*)carve((size_t)NN * 512 * 4);  // [we1_a|we1_b] projection
    float* zbias = (float*)carve(512 * 4);               // zero bias vector
    bf16 *we1abT[2], *we2T[2], *wn1T[2], *wn2T[2];
    float* wcd[2];
    for (int l = 0; l < 2; ++l) {
        we1abT[l] = (bf16*)carve(512 * 128 * 2);
        we2T[l]   = (bf16*)carve(128 * 256 * 2);
        wn1T[l]   = (bf16*)carve(256 * 256 * 2);
        wn2T[l]   = (bf16*)carve(128 * 256 * 2);
        wcd[l]    = (float*)carve(18 * 256 * 4);
    }
    bf16* injwT = (bf16*)carve(128 * 128 * 2);

    const int TB = 256;
    auto grid1 = [&](int n) { return (n + TB - 1) / TB; };

    // ---- one-time prep (re-run every call: stateless/deterministic) ----
    k_zero<<<grid1(NN * 128), TB, 0, stream>>>(zZ, NN * 128);
    k_zero<<<1, 512, 0, stream>>>(zbias, 512);
    k_transpose_bf16<<<grid1(128 * 128), TB, 0, stream>>>(inj_w, injwT, 128, 128);
    for (int l = 0; l < 2; ++l) {
        k_build_we1ab_t<<<grid1(512 * 128), TB, 0, stream>>>(W[l][0], we1abT[l]);
        k_build_wcd<<<grid1(18 * 256), TB, 0, stream>>>(W[l][0], W[l][1], wcd[l]);
        k_transpose_bf16<<<grid1(256 * 128), TB, 0, stream>>>(W[l][2], we2T[l], 256, 128);
        k_transpose_bf16<<<grid1(256 * 256), TB, 0, stream>>>(W[l][4], wn1T[l], 256, 256);
        k_transpose_bf16<<<grid1(256 * 128), TB, 0, stream>>>(W[l][6], wn2T[l], 256, 128);
    }
    // xin = x @ inj_w + inj_b
    k_gemm_n16k128<128><<<NN / 16, 128, 0, stream>>>(x, injwT, inj_b, xin);

    // ---- 8 Picard iterations, 2 EGNN layers each ----
    for (int it = 0; it < NITER; ++it) {
        // layer 1: z1 = egnn1(z0) ; z2 = gelu(z1 + xin)
        k_gemm_n16k128<512><<<NN / 16, 128, 0, stream>>>(zZ, we1abT[0], zbias, zw);
        k_zero<<<grid1(NN * 128), TB, 0, stream>>>(agg, NN * 128);
        k_edge<<<EE / 32, 256, 0, stream>>>(zw, pos, srcI, dstI, eattr,
                                            wcd[0], we2T[0], W[0][3], agg);
        k_node<<<NN / 16, 128, 0, stream>>>(zZ, agg, wn1T[0], W[0][5],
                                            wn2T[0], W[0][7], xin, z2);
        // layer 2: z3 = egnn2(z2) ; z_next = gelu(z3 + z0)
        k_gemm_n16k128<512><<<NN / 16, 128, 0, stream>>>(z2, we1abT[1], zbias, zw);
        k_zero<<<grid1(NN * 128), TB, 0, stream>>>(agg, NN * 128);
        k_edge<<<EE / 32, 256, 0, stream>>>(zw, pos, srcI, dstI, eattr,
                                            wcd[1], we2T[1], W[1][3], agg);
        float* zo = (it == NITER - 1) ? (float*)d_out : zZ;
        k_node<<<NN / 16, 128, 0, stream>>>(z2, agg, wn1T[1], W[1][5],
                                            wn2T[1], W[1][7], zZ, zo);
    }
}
